// PhiOnlyLoss_19181323944328
// MI455X (gfx1250) — matrix operand order repaired and checked by reading, not compiled, not run on hardware
//
#include <hip/hip_runtime.h>

// Problem constants (match reference setup_inputs)
#define M_  2
#define L_  6
#define N_  512
#define H_  64
#define V_K 1.0f
#define SIGMA2 0.01f
#define EPS_ 1e-10f

typedef __attribute__((ext_vector_type(16))) _Float16 v16h;
typedef __attribute__((ext_vector_type(8)))  float    v8f;

static __device__ __forceinline__ v8f wmma16(v16h a, v16h b, v8f c) {
  // D = A(16x32 f16) * B(32x16 f16) + C(16x16 f32)
  return __builtin_amdgcn_wmma_f32_16x16x32_f16(
      /*neg_a=*/false, a, /*neg_b=*/false, b,
      /*c_mod=*/(short)0, c, /*reuse_a=*/false, /*reuse_b=*/false);
}

static __device__ __forceinline__ float tanh_fast(float x) {
  float z = fminf(fmaxf(x, -15.f), 15.f);
  float e = __expf(2.f * z);
  return (e - 1.f) * __builtin_amdgcn_rcpf(e + 1.f);
}

static __device__ __forceinline__ float bfly16(float v) {
  v += __shfl_xor(v, 1, 32);
  v += __shfl_xor(v, 2, 32);
  v += __shfl_xor(v, 4, 32);
  v += __shfl_xor(v, 8, 32);
  return v;  // sum over the 16 lanes of this half
}
static __device__ __forceinline__ float bfly32(float v) {
  v = bfly16(v);
  v += __shfl_xor(v, 16, 32);
  return v;  // sum over all 32 lanes
}

// ---- W2 into WMMA 16-bit B 32x16 register layout -------------------------
// lanes 0-15 hold K = kt*32 + 0..15, lanes 16-31 hold K = kt*32 + 16..31,
// column N = nt*16 + lcol.
static __device__ __forceinline__ void load_B(const float* __restrict__ W2,
                                              int half, int lcol, v16h Bm[4][2]) {
#pragma unroll
  for (int nt = 0; nt < 4; ++nt) {
#pragma unroll
    for (int kt = 0; kt < 2; ++kt) {
      v16h bv;
#pragma unroll
      for (int e = 0; e < 16; ++e) {
        int h   = kt * 32 + half * 16 + e;
        int col = nt * 16 + lcol;
        bv[e] = (_Float16)W2[h * H_ + col];
      }
      Bm[nt][kt] = bv;
    }
  }
}

// A 16x32 f16 layout: this lane's element e maps to
//   k = kt*32 + 16*half + e
static __device__ __forceinline__ int a_kidx(int kt, int e, int half) {
  return kt * 32 + 16 * half + e;
}

// ============================================================================
// Kernel 1: gradient path, one wave32 per (m, l, i) with l in [0, L-2].
// Per 16-j chunk: three [16x64]x[64x64] mat-vecs (h1, dh1, d2h1 vs W2) as
// 24 straight-line v_wmma_f32_16x16x32_f16 with full EXEC.
// ============================================================================
__global__ __launch_bounds__(128) __attribute__((amdgpu_waves_per_eu(4)))
void phi_grad_wmma_kernel(
    const float* __restrict__ data, const float* __restrict__ t,
    const float* __restrict__ w1, const float* __restrict__ b1,
    const float* __restrict__ W2, const float* __restrict__ b2,
    const float* __restrict__ w3, const float* __restrict__ b3,
    float* __restrict__ partial)
{
  const int lane = threadIdx.x & 31;
  const int half = lane >> 4;
  const int lcol = lane & 15;
  const int wid  = blockIdx.x * (blockDim.x >> 5) + (threadIdx.x >> 5);
  const int i  = wid % N_;
  const int ml = wid / N_;
  const int l  = ml % (L_ - 1);
  const int m  = ml / (L_ - 1);
  const bool needPhi = (l == 0);       // telescoped energy: phi matters at l=0 only

  v16h Bm[4][2];
  load_B(W2, half, lcol, Bm);

  // per-lane w1/b1 slices in A-layout k order, packed f16 (A is f16 anyway)
  v16h w1h[2], b1h[2];
#pragma unroll
  for (int kt = 0; kt < 2; ++kt)
#pragma unroll
    for (int e = 0; e < 16; ++e) {
      int k = a_kidx(kt, e, half);
      w1h[kt][e] = (_Float16)w1[k];
      b1h[kt][e] = (_Float16)b1[k];
    }
  float b2r[4], w3r[4];
#pragma unroll
  for (int nt = 0; nt < 4; ++nt) {
    b2r[nt] = b2[nt * 16 + lcol];
    w3r[nt] = w3[nt * 16 + lcol];
  }

  const float* dml = data + (size_t)(m * L_ + l) * N_ * 2;
  const float xi = dml[i * 2 + 0];
  const float yi = dml[i * 2 + 1];

  float driftx = 0.f, drifty = 0.f, lap = 0.f, phis = 0.f;

  for (int j0 = 0; j0 < N_; j0 += 16) {
    const int   j   = j0 + lcol;
    const float dx  = xi - dml[j * 2 + 0];
    const float dy  = yi - dml[j * 2 + 1];
    const float ssq = dx * dx + dy * dy;
    const float r   = (ssq > 0.f) ? sqrtf(ssq) : 0.f;

    // h1 = tanh(r*w1+b1), dh1 = (1-h1^2)w1, d2h1 = -2 h1 dh1 w1
    v16h A1[2], Ad[2], Ad2[2];
#pragma unroll
    for (int kt = 0; kt < 2; ++kt)
#pragma unroll
      for (int e = 0; e < 16; ++e) {
        float w   = (float)w1h[kt][e];
        float h1  = tanh_fast(r * w + (float)b1h[kt][e]);
        float g   = 1.f - h1 * h1;
        float dh1 = g * w;
        A1[kt][e]  = (_Float16)h1;
        Ad[kt][e]  = (_Float16)dh1;
        Ad2[kt][e] = (_Float16)(-2.f * h1 * dh1 * w);
      }

    const bool inCh = (i >= j0) && (i < j0 + 16);   // wave-uniform
    const int  ri   = i - j0;

    float sp = 0.f, sl = 0.f;       // chunk totals (phi, laplacian partials)
    float vp = 0.f, vl = 0.f;       // diagonal (j==i) element partials
    float pdphi[8] = {};            // per-row dphi partials (needed for drift)
#pragma unroll
    for (int nt = 0; nt < 4; ++nt) {
      // three independent accumulate chains, interleaved per shared B operand
      v8f a = {}, da = {}, d2a = {};
      a   = wmma16(A1[0],  Bm[nt][0], a);
      da  = wmma16(Ad[0],  Bm[nt][0], da);
      d2a = wmma16(Ad2[0], Bm[nt][0], d2a);
      a   = wmma16(A1[1],  Bm[nt][1], a);
      da  = wmma16(Ad[1],  Bm[nt][1], da);
      d2a = wmma16(Ad2[1], Bm[nt][1], d2a);

      const float b2k = b2r[nt], w3k = w3r[nt];
#pragma unroll
      for (int p = 0; p < 8; ++p) {
        float h2   = tanh_fast(a[p] + b2k);
        float phik = h2 * w3k;
        sp += phik;
        float g2   = 1.f - h2 * h2;
        float dav  = da[p];
        float dh2  = g2 * dav;
        float lapk = (-2.f * h2 * dh2 * dav + g2 * d2a[p]) * w3k;
        sl += lapk;
        pdphi[p] += dh2 * w3k;
        if (inCh && ((ri & 7) == p)) { vp += phik; vl += lapk; }
      }
    }

    // exclude diagonal (j == i) from phi / laplacian sums
    if (inCh) {
      float bp = bfly16(vp);
      float bl = bfly16(vl);
      int src = (ri >> 3) << 4;     // lane 0 or 16: owner half of row ri
      if (needPhi) phis -= __shfl(bp, src, 32);
      lap -= __shfl(bl, src, 32);
    }
    if (needPhi) phis += bfly32(sp);
    lap += bfly32(sl);

    // drift: sum_j dphi(r_ij) * diff_ij / max(r, EPS)  (diag auto-zero)
    float cx = 0.f, cy = 0.f;
#pragma unroll
    for (int p = 0; p < 8; ++p) {
      float rs  = bfly16(pdphi[p]);        // row sum for row 8*half + p
      int   src = half * 8 + p;            // lane that cached that row's diff/r
      float dxr = __shfl(dx, src, 32);
      float dyr = __shfl(dy, src, 32);
      float rr  = __shfl(r,  src, 32);
      float inv = __builtin_amdgcn_rcpf(fmaxf(rr, EPS_));
      cx += rs * dxr * inv;
      cy += rs * dyr * inv;
    }
    driftx += cx + __shfl_xor(cx, 16, 32);
    drifty += cy + __shfl_xor(cy, 16, 32);
  }

  const float invN = 1.f / (float)N_;
  float dfx = -V_K * xi - driftx * invN;
  float dfy = -V_K * yi - drifty * invN;
  float dtl = t[l + 1] - t[l];
  float contrib = dtl * ((dfx * dfx + dfy * dfy) * invN + SIGMA2 * lap * invN * invN);
  if (needPhi) {  // energy at l=0 enters with weight -1 -> contrib -= 2*(-1)*E_i
    float phi_i = phis + (float)(N_ - 1) * b3[0];
    contrib += 2.f * (0.5f * V_K * (xi * xi + yi * yi) * invN + phi_i * invN * invN);
  }
  if (lane == 0) partial[wid] = contrib;
}

// ============================================================================
// Kernel 2: phi-only path for l = L-1 (energy endpoint, weight +1).
// One mat-vec per chunk (8 WMMAs), no derivative state -> low VGPR pressure.
// ============================================================================
__global__ __launch_bounds__(128) void phi_only_wmma_kernel(
    const float* __restrict__ data,
    const float* __restrict__ w1, const float* __restrict__ b1,
    const float* __restrict__ W2, const float* __restrict__ b2,
    const float* __restrict__ w3, const float* __restrict__ b3,
    float* __restrict__ partial)
{
  const int lane = threadIdx.x & 31;
  const int half = lane >> 4;
  const int lcol = lane & 15;
  const int wid  = blockIdx.x * (blockDim.x >> 5) + (threadIdx.x >> 5);
  const int i = wid % N_;
  const int m = wid / N_;
  const int l = L_ - 1;

  v16h Bm[4][2];
  load_B(W2, half, lcol, Bm);

  v16h w1h[2], b1h[2];
#pragma unroll
  for (int kt = 0; kt < 2; ++kt)
#pragma unroll
    for (int e = 0; e < 16; ++e) {
      int k = a_kidx(kt, e, half);
      w1h[kt][e] = (_Float16)w1[k];
      b1h[kt][e] = (_Float16)b1[k];
    }
  float b2r[4], w3r[4];
#pragma unroll
  for (int nt = 0; nt < 4; ++nt) {
    b2r[nt] = b2[nt * 16 + lcol];
    w3r[nt] = w3[nt * 16 + lcol];
  }

  const float* dml = data + (size_t)(m * L_ + l) * N_ * 2;
  const float xi = dml[i * 2 + 0];
  const float yi = dml[i * 2 + 1];

  float phis = 0.f;

  for (int j0 = 0; j0 < N_; j0 += 16) {
    const int   j   = j0 + lcol;
    const float dx  = xi - dml[j * 2 + 0];
    const float dy  = yi - dml[j * 2 + 1];
    const float ssq = dx * dx + dy * dy;
    const float r   = (ssq > 0.f) ? sqrtf(ssq) : 0.f;

    v16h A1[2];
#pragma unroll
    for (int kt = 0; kt < 2; ++kt)
#pragma unroll
      for (int e = 0; e < 16; ++e)
        A1[kt][e] = (_Float16)tanh_fast(r * (float)w1h[kt][e] + (float)b1h[kt][e]);

    const bool inCh = (i >= j0) && (i < j0 + 16);
    const int  ri   = i - j0;
    float sp = 0.f, vp = 0.f;
#pragma unroll
    for (int nt = 0; nt < 4; ++nt) {
      v8f a = {};
      a = wmma16(A1[0], Bm[nt][0], a);
      a = wmma16(A1[1], Bm[nt][1], a);
      const float b2k = b2r[nt], w3k = w3r[nt];
#pragma unroll
      for (int p = 0; p < 8; ++p) {
        float phik = tanh_fast(a[p] + b2k) * w3k;
        sp += phik;
        if (inCh && ((ri & 7) == p)) vp += phik;
      }
    }

    if (inCh) {                      // exclude diagonal
      float bp = bfly16(vp);
      phis -= __shfl(bp, (ri >> 3) << 4, 32);
    }
    phis += bfly32(sp);
  }

  // energy at l=L-1 enters with weight +1 -> contrib = -2*(+1)*E_i
  const float invN = 1.f / (float)N_;
  float phi_i = phis + (float)(N_ - 1) * b3[0];
  float contrib = -2.f * (0.5f * V_K * (xi * xi + yi * yi) * invN + phi_i * invN * invN);
  if (lane == 0) partial[M_ * (L_ - 1) * N_ + wid] = contrib;
}

// Deterministic fixed-order reduction of all partials -> residual^2
__global__ __launch_bounds__(256) void phi_reduce_kernel(
    const float* __restrict__ partial, const float* __restrict__ t,
    float* __restrict__ out)
{
  __shared__ float sm[256];
  float s = 0.f;
  for (int idx = threadIdx.x; idx < M_ * L_ * N_; idx += 256) s += partial[idx];
  sm[threadIdx.x] = s;
  __syncthreads();
  for (int off = 128; off > 0; off >>= 1) {
    if ((int)threadIdx.x < off) sm[threadIdx.x] += sm[threadIdx.x + off];
    __syncthreads();
  }
  if (threadIdx.x == 0) {
    float S = sm[0];
    float dts = 0.f;
    for (int l = 0; l < L_ - 1; ++l) dts += t[l + 1] - t[l];
    S += SIGMA2 * V_K * 2.0f /* d */ * dts * (float)M_;  // constant diffusion term
    float residual = S / (float)(M_ * (L_ - 1));         // n_pairs = M*(L-1)
    out[0] = residual * residual;
  }
}

extern "C" void kernel_launch(void* const* d_in, const int* in_sizes, int n_in,
                              void* d_out, int out_size, void* d_ws, size_t ws_size,
                              hipStream_t stream) {
  (void)in_sizes; (void)n_in; (void)out_size; (void)ws_size;
  const float* data = (const float*)d_in[0];
  const float* t    = (const float*)d_in[1];
  const float* w1   = (const float*)d_in[2];
  const float* b1   = (const float*)d_in[3];
  const float* W2   = (const float*)d_in[4];
  const float* b2   = (const float*)d_in[5];
  const float* w3   = (const float*)d_in[6];
  const float* b3   = (const float*)d_in[7];
  float* out     = (float*)d_out;
  float* partial = (float*)d_ws;   // M*L*N = 6144 floats (24 KB): [grad | phi]

  const int wavesPB = 4;                              // 128-thread blocks
  const int nGrad   = M_ * (L_ - 1) * N_;             // 5120 waves
  const int nPhi    = M_ * N_;                        // 1024 waves

  phi_grad_wmma_kernel<<<nGrad / wavesPB, 32 * wavesPB, 0, stream>>>(
      data, t, w1, b1, W2, b2, w3, b3, partial);
  phi_only_wmma_kernel<<<nPhi / wavesPB, 32 * wavesPB, 0, stream>>>(
      data, w1, b1, W2, b2, w3, b3, partial);
  phi_reduce_kernel<<<1, 256, 0, stream>>>(partial, t, out);
}
